// MetaAttention_76871324664379
// MI455X (gfx1250) — matrix operand order repaired and checked by reading, not compiled, not run on hardware
//
#include <hip/hip_runtime.h>
#include <hip/hip_bf16.h>

typedef __attribute__((ext_vector_type(16))) _Float16 v16h;
typedef __attribute__((ext_vector_type(8)))  _Float16 v8h;
typedef __attribute__((ext_vector_type(8)))  float    v8f;

namespace cfg {
constexpr int B = 2, T = 2048, C = 2048, H = 16, D = 128, MEM = 1024, S = T + MEM;
constexpr float SCALE = 0.08838834764831845f;   // 1/sqrt(D)
constexpr float L2E   = 1.4426950408889634f;    // log2(e)
}

// ---------------------------------------------------------------- helpers ---
__device__ __forceinline__ v16h ld_frag(const _Float16* p0, const _Float16* p1) {
    union { v16h v; v8h h[2]; } u;
    u.h[0] = *(const v8h*)p0;
    u.h[1] = *(const v8h*)p1;
    return u.v;
}

__device__ __forceinline__ v8f wmma_f16(v16h a, v16h b, v8f c) {
    // D = A(16x32 f16) * B(32x16 f16) + C(16x16 f32)
    return __builtin_amdgcn_wmma_f32_16x16x32_f16(false, a, false, b,
                                                  (short)0, c, false, false);
}

// ------------------------------------------------------- convert / reshape ---
__global__ void convert_f32_to_f16(const float* __restrict__ in,
                                   _Float16* __restrict__ out, long n) {
    long i = (long)blockIdx.x * blockDim.x + threadIdx.x;
    if (i < n) out[i] = (_Float16)in[i];
}

// in: f32 [R][Cc] row-major  ->  out: f16 [Cc][R] row-major
__global__ void transpose_cvt(const float* __restrict__ in,
                              _Float16* __restrict__ out, int R, int Cc) {
    __shared__ _Float16 tile[32][36];
    const int bx = blockIdx.x * 32;           // column base (Cc dim)
    const int by = blockIdx.y * 32;           // row base    (R dim)
    const int x = threadIdx.x, y = threadIdx.y;
#pragma unroll
    for (int i = 0; i < 4; ++i) {
        int r = by + y + i * 8;
        tile[y + i * 8][x] = (_Float16)in[(size_t)r * Cc + bx + x];
    }
    __syncthreads();
#pragma unroll
    for (int i = 0; i < 4; ++i) {
        int c = bx + y + i * 8;
        out[(size_t)c * R + by + x] = tile[x][y + i * 8];
    }
}

// copy memory bank (f32) into tail rows [T, S) of k/v f16 buffers, all batches
__global__ void fill_mem_tails(const float* __restrict__ mem,
                               _Float16* __restrict__ k, _Float16* __restrict__ v) {
    using namespace cfg;
    long i = (long)blockIdx.x * blockDim.x + threadIdx.x;     // over B*MEM*C
    if (i >= (long)B * MEM * C) return;
    int c  = (int)(i % C);
    long r = i / C;
    int m  = (int)(r % MEM);
    int b  = (int)(r / MEM);
    _Float16 hv = (_Float16)mem[(size_t)m * C + c];
    size_t o = ((size_t)b * S + T + m) * C + c;
    k[o] = hv;
    v[o] = hv;
}

// vbuf f16 [B][S][C] -> vt f16 [B][H][D][S]
__global__ void transpose_v(const _Float16* __restrict__ vbuf,
                            _Float16* __restrict__ vt) {
    using namespace cfg;
    __shared__ _Float16 tile[32][36];
    const int bh = blockIdx.z;
    const int b = bh / H, h = bh % H;
    const _Float16* in = vbuf + (size_t)b * S * C + h * D;
    _Float16* out = vt + (size_t)bh * D * S;
    const int s0 = blockIdx.x * 32, d0 = blockIdx.y * 32;
    const int x = threadIdx.x, y = threadIdx.y;
#pragma unroll
    for (int i = 0; i < 4; ++i)
        tile[y + i * 8][x] = in[(size_t)(s0 + y + i * 8) * C + d0 + x];
    __syncthreads();
#pragma unroll
    for (int i = 0; i < 4; ++i)
        out[(size_t)(d0 + y + i * 8) * S + s0 + x] = tile[x][y + i * 8];
}

// --------------------------------------- ping-pong software-pipelined GEMM ---
// One wave computes a 32(M) x 64(N) tile: acc[mi][c], mi in {0,1}, c in 0..3.
// Two fragment generations alternate roles -> no register-rotation moves.
struct Gen { v16h a0, a1, b[4]; };

__device__ __forceinline__ void load_gen(Gen& g, const _Float16* arow0,
                                         const _Float16* arow1,
                                         const _Float16* const brow[4], int kc) {
    g.a0 = ld_frag(arow0 + kc, arow0 + kc + 16);
    g.a1 = ld_frag(arow1 + kc, arow1 + kc + 16);
#pragma unroll
    for (int c = 0; c < 4; ++c) g.b[c] = ld_frag(brow[c] + kc, brow[c] + kc + 16);
}

__device__ __forceinline__ void comp_gen(const Gen& g, v8f acc[2][4]) {
#pragma unroll
    for (int c = 0; c < 4; ++c) {
        acc[0][c] = wmma_f16(g.a0, g.b[c], acc[0][c]);
        acc[1][c] = wmma_f16(g.a1, g.b[c], acc[1][c]);
    }
}

__device__ __forceinline__ void gemm_tile_32x64(
    const _Float16* __restrict__ A,   // [Mrows][C] row-major (f16)
    const _Float16* __restrict__ Bt,  // [N][C]   row-major (f16, pre-transposed)
    int m0, int ntile, int lane, v8f acc[2][4]) {
    using namespace cfg;
    const int n16 = lane & 15, half = lane >> 4;
    const _Float16* arow0 = A + (size_t)(m0 + n16) * C + half * 8;
    const _Float16* arow1 = arow0 + (size_t)16 * C;
    const _Float16* brow[4];
#pragma unroll
    for (int c = 0; c < 4; ++c)
        brow[c] = Bt + (size_t)(ntile + c * 16 + n16) * C + half * 8;

    Gen gA, gB;
    load_gen(gA, arow0, arow1, brow, 0);
#pragma unroll 1
    for (int kc = 0; kc < C - 64; kc += 64) {
        __builtin_prefetch(arow0 + kc + 512, 0, 0);
        load_gen(gB, arow0, arow1, brow, kc + 32);
        comp_gen(gA, acc);
        load_gen(gA, arow0, arow1, brow, kc + 64);
        comp_gen(gB, acc);
    }
    load_gen(gB, arow0, arow1, brow, C - 32);
    comp_gen(gA, acc);    // k-step C-64
    comp_gen(gB, acc);    // k-step C-32
}

// qkv = xh(f16)[B*T][C] @ W^T(f16)[3C][C] + bias -> split f16 q/k/v
__global__ __launch_bounds__(128)
void gemm_qkv(const _Float16* __restrict__ xh, const _Float16* __restrict__ wT,
              const float* __restrict__ bias, _Float16* __restrict__ q,
              _Float16* __restrict__ k, _Float16* __restrict__ v) {
    using namespace cfg;
    const int lane = threadIdx.x & 31, wave = threadIdx.x >> 5;
    const int n16 = lane & 15, half = lane >> 4;
    const int m0 = blockIdx.y * 32;
    const int ntile = blockIdx.x * 256 + wave * 64;
    v8f zero = {};
    v8f acc[2][4] = {{zero, zero, zero, zero}, {zero, zero, zero, zero}};
    gemm_tile_32x64(xh, wT, m0, ntile, lane, acc);
#pragma unroll
    for (int c = 0; c < 4; ++c) {
        const int n = ntile + c * 16 + n16;
        const float bval = bias[n];
#pragma unroll
        for (int mi = 0; mi < 2; ++mi)
#pragma unroll
            for (int j = 0; j < 8; ++j) {
                int gm = m0 + mi * 16 + j + half * 8;   // row in [0, B*T)
                int bb = gm / T, tt = gm % T;
                _Float16 hv = (_Float16)(acc[mi][c][j] + bval);
                if (n < C)          q[(size_t)gm * C + n] = hv;
                else if (n < 2 * C) k[((size_t)bb * S + tt) * C + (n - C)] = hv;
                else                v[((size_t)bb * S + tt) * C + (n - 2 * C)] = hv;
            }
    }
}

// out(f32) = hid(f16)[B*T][C] @ Wp^T(f16)[C][C] + b_proj
__global__ __launch_bounds__(128)
void gemm_proj(const _Float16* __restrict__ hid, const _Float16* __restrict__ wT,
               const float* __restrict__ bias, float* __restrict__ out) {
    using namespace cfg;
    const int lane = threadIdx.x & 31, wave = threadIdx.x >> 5;
    const int n16 = lane & 15, half = lane >> 4;
    const int m0 = blockIdx.y * 32;
    const int ntile = blockIdx.x * 256 + wave * 64;
    v8f zero = {};
    v8f acc[2][4] = {{zero, zero, zero, zero}, {zero, zero, zero, zero}};
    gemm_tile_32x64(hid, wT, m0, ntile, lane, acc);
#pragma unroll
    for (int c = 0; c < 4; ++c) {
        const int n = ntile + c * 16 + n16;
        const float bval = bias[n];
#pragma unroll
        for (int mi = 0; mi < 2; ++mi)
#pragma unroll
            for (int j = 0; j < 8; ++j) {
                int gm = m0 + mi * 16 + j + half * 8;
                out[(size_t)gm * C + n] = acc[mi][c][j] + bval;
            }
    }
}

// --------------------------------------------------------------- attention ---
__device__ __forceinline__ void attn_chunk(
    const _Float16* __restrict__ kbase,   // kbuf + b*S*C + h*D
    const _Float16* __restrict__ vtbase,  // vt + (b*H+h)*D*S
    _Float16* __restrict__ pl,            // wave-private LDS, 16 rows * stride 40
    const v16h* __restrict__ aq,          // 4 Q A-fragments (D=128)
    int s0, int qt0, bool masked,
    v8f* __restrict__ acc, float* __restrict__ mrow, float* __restrict__ lrow,
    int lane) {
    using namespace cfg;
    const int n16 = lane & 15, half = lane >> 4;
    // --- load all K B-fragments for both 16-key tiles, then 8 wmmas ----------
    v16h bk[2][4];
#pragma unroll
    for (int t = 0; t < 2; ++t) {
        const _Float16* kr = kbase + (size_t)(s0 + t * 16 + n16) * C + half * 8;
#pragma unroll
        for (int c0 = 0; c0 < 4; ++c0)
            bk[t][c0] = ld_frag(kr + c0 * 32, kr + c0 * 32 + 16);
    }
    v8f s[2];
#pragma unroll
    for (int t = 0; t < 2; ++t) {
        v8f st = {};
#pragma unroll
        for (int c0 = 0; c0 < 4; ++c0) st = wmma_f16(aq[c0], bk[t][c0], st);
        s[t] = st;
    }
    // --- issue V B-fragment loads now; they fly during the softmax VALU ------
    v16h bv[8];
#pragma unroll
    for (int c = 0; c < 8; ++c) {
        const _Float16* vp = vtbase + (size_t)(c * 16 + n16) * S + s0 + half * 8;
        bv[c] = ld_frag(vp, vp + 16);
    }
    // --- online softmax ------------------------------------------------------
#pragma unroll
    for (int j = 0; j < 8; ++j) {
        float s0v = s[0][j] * SCALE, s1v = s[1][j] * SCALE;
        if (masked) {
            int qi = qt0 + j + half * 8;
            if (s0 + n16 > qi)      s0v = -1e30f;   // causal region only
            if (s0 + 16 + n16 > qi) s1v = -1e30f;
        }
        float t0 = fmaxf(s0v, s1v);
        t0 = fmaxf(t0, __shfl_xor(t0, 1, 32));
        t0 = fmaxf(t0, __shfl_xor(t0, 2, 32));
        t0 = fmaxf(t0, __shfl_xor(t0, 4, 32));
        t0 = fmaxf(t0, __shfl_xor(t0, 8, 32));
        float mnew = fmaxf(mrow[j], t0);
        float sc = exp2f((mrow[j] - mnew) * L2E);
        float p0 = exp2f((s0v - mnew) * L2E);
        float p1 = exp2f((s1v - mnew) * L2E);
        float ps = p0 + p1;
        ps += __shfl_xor(ps, 1, 32);
        ps += __shfl_xor(ps, 2, 32);
        ps += __shfl_xor(ps, 4, 32);
        ps += __shfl_xor(ps, 8, 32);
        lrow[j] = lrow[j] * sc + ps;
        mrow[j] = mnew;
#pragma unroll
        for (int c = 0; c < 8; ++c) acc[c][j] *= sc;
        const int r = j + half * 8;
        pl[r * 40 + n16]      = (_Float16)p0;
        pl[r * 40 + 16 + n16] = (_Float16)p1;
    }
    // wave-private LDS RAW: DS ops are in-order within a wave; wait for safety
    asm volatile("s_wait_dscnt 0x0" ::: "memory");
    // --- P (16x32) as A-fragment x preloaded V fragments ---------------------
    v16h pa = ld_frag(pl + n16 * 40 + half * 8, pl + n16 * 40 + half * 8 + 16);
#pragma unroll
    for (int c = 0; c < 8; ++c) acc[c] = wmma_f16(pa, bv[c], acc[c]);
}

__global__ __launch_bounds__(128)
void attn_kernel(const _Float16* __restrict__ qbuf, const _Float16* __restrict__ kbuf,
                 const _Float16* __restrict__ vt, _Float16* __restrict__ hid) {
    using namespace cfg;
    __shared__ _Float16 plds[4][16 * 40];
    const int lane = threadIdx.x & 31, wave = threadIdx.x >> 5;
    const int n16 = lane & 15, half = lane >> 4;
    const int bh = blockIdx.x / (T / 64);
    const int tc = blockIdx.x % (T / 64);
    const int b = bh / H, h = bh % H;
    const int qt0 = tc * 64 + wave * 16;

    const _Float16* qrow = qbuf + (size_t)(b * T + qt0 + n16) * C + h * D + half * 8;
    v16h aq[4];
#pragma unroll
    for (int c0 = 0; c0 < 4; ++c0)
        aq[c0] = ld_frag(qrow + c0 * 32, qrow + c0 * 32 + 16);

    v8f zero = {};
    v8f acc[8];
    float mrow[8], lrow[8];
#pragma unroll
    for (int j = 0; j < 8; ++j) { acc[j] = zero; mrow[j] = -3.0e38f; lrow[j] = 0.f; }

    const _Float16* kbase = kbuf + (size_t)b * S * C + h * D;
    const _Float16* vtbase = vt + (size_t)(b * H + h) * D * S;
    _Float16* pl = plds[wave];

    const int nc = qt0 / 32 + 1;                       // causal 32-key chunks
    for (int i = 0; i < nc; ++i)
        attn_chunk(kbase, vtbase, pl, aq, i * 32, qt0, i == nc - 1,
                   acc, mrow, lrow, lane);
    for (int i = 0; i < MEM / 32; ++i)                 // memory-bank chunks
        attn_chunk(kbase, vtbase, pl, aq, T + i * 32, qt0, false,
                   acc, mrow, lrow, lane);

#pragma unroll
    for (int c = 0; c < 8; ++c)
#pragma unroll
        for (int j = 0; j < 8; ++j) {
            int qi = qt0 + j + half * 8;
            hid[(size_t)(b * T + qi) * C + h * D + c * 16 + n16] =
                (_Float16)(acc[c][j] / lrow[j]);
        }
}

// ------------------------------------------------------------------- launch ---
extern "C" void kernel_launch(void* const* d_in, const int* in_sizes, int n_in,
                              void* d_out, int out_size, void* d_ws, size_t ws_size,
                              hipStream_t stream) {
    using namespace cfg;
    const float* x      = (const float*)d_in[0];
    const float* W_qkv  = (const float*)d_in[1];
    const float* b_qkv  = (const float*)d_in[2];
    const float* memb   = (const float*)d_in[3];
    const float* W_proj = (const float*)d_in[4];
    const float* b_proj = (const float*)d_in[5];
    float* out = (float*)d_out;

    char* p = (char*)d_ws;
    const size_t szXH = (size_t)B * T * C * 2;      // x in f16
    const size_t szWQ = (size_t)3 * C * C * 2;      // W_qkv^T f16
    const size_t szWP = (size_t)C * C * 2;          // W_proj^T f16
    const size_t szQ  = (size_t)B * T * C * 2;      // q f16
    const size_t szKV = (size_t)B * S * C * 2;      // k / v f16
    const size_t szH  = (size_t)B * T * C * 2;      // attention hidden f16
    const size_t need = szXH + szWQ + szWP + szQ + 3 * szKV + szH;
    if (ws_size < need) return;

    _Float16* xh    = (_Float16*)p;            p += szXH;
    _Float16* wqkvT = (_Float16*)p;            p += szWQ;
    _Float16* wprojT= (_Float16*)p;            p += szWP;
    _Float16* qb    = (_Float16*)p;            p += szQ;
    _Float16* kb    = (_Float16*)p;            p += szKV;
    _Float16* vb    = (_Float16*)p;            p += szKV;
    _Float16* vt    = (_Float16*)p;            p += szKV;   // [B][H][D][S]
    _Float16* hid   = (_Float16*)p;            p += szH;

    const long nX = (long)B * T * C;
    convert_f32_to_f16<<<(nX + 255) / 256, 256, 0, stream>>>(x, xh, nX);
    transpose_cvt<<<dim3(3 * C / 32, C / 32), dim3(32, 8), 0, stream>>>(W_qkv, wqkvT, C, 3 * C);
    transpose_cvt<<<dim3(C / 32, C / 32), dim3(32, 8), 0, stream>>>(W_proj, wprojT, C, C);
    gemm_qkv<<<dim3(3 * C / 256, B * T / 32), 128, 0, stream>>>(xh, wqkvT, b_qkv, qb, kb, vb);
    const long nM = (long)B * MEM * C;
    fill_mem_tails<<<(nM + 255) / 256, 256, 0, stream>>>(memb, kb, vb);
    transpose_v<<<dim3(S / 32, D / 32, B * H), dim3(32, 8), 0, stream>>>(vb, vt);
    attn_kernel<<<B * H * (T / 64), 128, 0, stream>>>(qb, kb, vt, hid);
    gemm_proj<<<dim3(C / 256, B * T / 32), 128, 0, stream>>>(hid, wprojT, b_proj, out);
}
